// RTDetrTransformer_14499809591616
// MI455X (gfx1250) — matrix operand order repaired and checked by reading, not compiled, not run on hardware
//
#include <hip/hip_runtime.h>
#include <hip/hip_bf16.h>
#include <math.h>

typedef __attribute__((ext_vector_type(16))) __bf16 v16bf;
typedef __attribute__((ext_vector_type(8)))  float  v8f;

#define NUM_HEADS 8
#define LQ        300
#define DIM       256
#define BATCH     16
#define LV        8400
#define KDIM      256   // all GEMMs here have K = 256

// ---------- helpers ----------
static __device__ __forceinline__ unsigned short f2bfbits(float f) {
  unsigned int u = __builtin_bit_cast(unsigned int, f);
  return (unsigned short)((u + 0x7FFFu + ((u >> 16) & 1u)) >> 16);  // RNE
}

// pack two fp32 -> one dword of two bf16 (round-half-away): 2x v_add + 1x v_perm_b32
static __device__ __forceinline__ unsigned int pk2bf(float a, float b) {
  unsigned int ua = __builtin_bit_cast(unsigned int, a) + 0x8000u;
  unsigned int ub = __builtin_bit_cast(unsigned int, b) + 0x8000u;
  return __builtin_amdgcn_perm(ub, ua, 0x07060302u);  // {ub.hi16, ua.hi16}
}

struct __align__(16) U32B { unsigned int w[8]; };  // 32 B, trivially copyable

// =====================================================================
// GEMM: Y[M,N] = X[M,K=256] * W[256,N] + bias[N]  (fp32 in/out, bf16 WMMA).
// Block = 256 thr = 8 waves (4 in M x 2 in N).  Workgroup M-tile = 128.
// A-slab (128 x 256) converted to bf16 ONCE into registers (aF[8][2]/wave),
// then the workgroup loops over N in 64-wide tiles: stage W-tile to LDS,
// 8 k-steps x 4 WMMA, store.  => X is read exactly once from HBM.
// =====================================================================
__global__ __launch_bounds__(256) void gemm_bias_wmma(
    const float* __restrict__ X, const float* __restrict__ W,
    const float* __restrict__ bias, float* __restrict__ Y,
    int M, int N)
{
  // W tile as bf16: ldsW[n][k]; row stride 272 ushorts = 544 B = 17*32
  // -> every (n, k%16==0) address is 32 B aligned for b128 LDS loads.
  __shared__ __align__(32) unsigned short ldsW[64][272];

  const int tid    = threadIdx.x;
  const int mbase  = blockIdx.y * 128;
  const int wave   = tid >> 5;
  const int lane   = tid & 31;
  const int wm     = wave >> 1;           // 0..3
  const int wn     = wave & 1;            // 0..1
  const int mtile  = mbase + wm * 32;
  const int laneLo = lane & 15;
  const int hiSel  = lane >> 4;           // 0|1
  const int kA     = hiSel * 16;          // A frag K base (contiguous K, matches B)
  const int kB     = hiSel * 16;          // B frag K base (contiguous K, matches A)

  // ---- load + convert the full-K A fragments once (rows clamped; stores guarded)
  v16bf aF[KDIM / 32][2];
#pragma unroll
  for (int ks = 0; ks < KDIM / 32; ++ks) {
#pragma unroll
    for (int i = 0; i < 2; ++i) {
      int row  = mtile + i * 16 + laneLo;
      int rrow = row < M ? row : M - 1;
      const float* xp = X + (size_t)rrow * KDIM + ks * 32 + kA;
      float4 f0 = *(const float4*)(xp);
      float4 f1 = *(const float4*)(xp + 4);
      float4 f2 = *(const float4*)(xp + 8);
      float4 f3 = *(const float4*)(xp + 12);
      U32B a;
      a.w[0] = pk2bf(f0.x, f0.y); a.w[1] = pk2bf(f0.z, f0.w);
      a.w[2] = pk2bf(f1.x, f1.y); a.w[3] = pk2bf(f1.z, f1.w);
      a.w[4] = pk2bf(f2.x, f2.y); a.w[5] = pk2bf(f2.z, f2.w);
      a.w[6] = pk2bf(f3.x, f3.y); a.w[7] = pk2bf(f3.z, f3.w);
      aF[ks][i] = __builtin_bit_cast(v16bf, a);
    }
  }

  const int NT = (N + 63) >> 6;     // number of 64-wide n-tiles (<= 4)
  const int rbase = hiSel * 8;

  for (int nt = 0; nt < NT; ++nt) {
    const int nbase = nt * 64;

    // ---- stage W[:, nbase:nbase+64] -> LDS as bf16 (coalesced over n)
    __syncthreads();   // protect previous tile's LDS reads
    for (int idx = tid; idx < 64 * KDIM; idx += 256) {
      int n = idx & 63;
      int k = idx >> 6;
      int col = nbase + n; if (col >= N) col = N - 1;   // clamp; stores guarded
      ldsW[n][k] = f2bfbits(W[(size_t)k * N + col]);
    }
    __syncthreads();

    // ---- 8 k-steps x (2x2) WMMA
    v8f acc[2][2] = {};
#pragma unroll
    for (int ks = 0; ks < KDIM / 32; ++ks) {
      v16bf bF[2];
#pragma unroll
      for (int j = 0; j < 2; ++j) {
        int ncol = wn * 32 + j * 16 + laneLo;           // 0..63
        U32B raw = *(const U32B*)&ldsW[ncol][ks * 32 + kB];
        bF[j] = __builtin_bit_cast(v16bf, raw);
      }
#pragma unroll
      for (int i = 0; i < 2; ++i)
#pragma unroll
        for (int j = 0; j < 2; ++j)
          acc[i][j] = __builtin_amdgcn_wmma_f32_16x16x32_bf16(
              false, aF[ks][i], false, bF[j], (short)0, acc[i][j], false, false);
    }

    // ---- store D (+bias): lane<16 -> rows r, lane>=16 -> rows 8+r; col = laneLo
#pragma unroll
    for (int i = 0; i < 2; ++i) {
#pragma unroll
      for (int j = 0; j < 2; ++j) {
        int col = nbase + wn * 32 + j * 16 + laneLo;
        if (col < N) {
          float bv = bias[col];
#pragma unroll
          for (int r = 0; r < 8; ++r) {
            int row = mtile + i * 16 + rbase + r;
            if (row < M) Y[(size_t)row * N + col] = acc[i][j][r] + bv;
          }
        }
      }
    }
  }
}

// =====================================================================
// Softmax over the 12 (level,point) logits per (b,q,head), in place.
// =====================================================================
__global__ void softmax12_kernel(float* __restrict__ attn, int total)
{
  int t = blockIdx.x * blockDim.x + threadIdx.x;
  if (t >= total) return;
  float* p = attn + (size_t)(t >> 3) * 96 + (size_t)(t & 7) * 12;
  float mx = p[0];
#pragma unroll
  for (int i = 1; i < 12; ++i) mx = fmaxf(mx, p[i]);
  float e[12], s = 0.0f;
#pragma unroll
  for (int i = 0; i < 12; ++i) { e[i] = __expf(p[i] - mx); s += e[i]; }
  float inv = 1.0f / s;
#pragma unroll
  for (int i = 0; i < 12; ++i) p[i] = e[i] * inv;
}

// =====================================================================
// Deformable sampling + weighted aggregation.
// One wave per (b,q,head); lane = channel (hd=32 == wave32).
// Each tap reads 32 contiguous fp32 channels -> one coalesced 128 B access.
// vproj (137 MB) ~fits the 192 MB L2, so taps are mostly L2 hits.
// =====================================================================
__global__ __launch_bounds__(256) void ms_deform_sample_kernel(
    const float* __restrict__ vproj,   // [B, LV, 256]
    const float* __restrict__ refp,    // [B, LQ, 4]
    const float* __restrict__ off,     // [B*LQ, 192]  (h,l,p,2)
    const float* __restrict__ attn,    // [B*LQ, 96]   (h,l,p) softmaxed
    float* __restrict__ out)           // [B*LQ, 256]
{
  const int lane = threadIdx.x & 31;
  const int widx = blockIdx.x * 8 + (threadIdx.x >> 5);   // 0..38399
  const int h  = widx & 7;
  const int bq = widx >> 3;
  const int b  = bq / LQ;

  const float4 rp   = *(const float4*)(refp + (size_t)bq * 4);
  const float* offp = off  + (size_t)bq * 192 + h * 24;
  const float* awp  = attn + (size_t)bq * 96  + h * 12;
  const float* vb   = vproj + (size_t)b * LV * DIM + h * 32 + lane;

  const int lw[3]     = {80, 40, 20};
  const int lh[3]     = {80, 40, 20};
  const int lstart[3] = {0, 6400, 8000};

  float acc = 0.0f;
#pragma unroll
  for (int l = 0; l < 3; ++l) {
    const int Wl = lw[l], Hl = lh[l], s0 = lstart[l];
#pragma unroll
    for (int p = 0; p < 4; ++p) {
      float ox = offp[(l * 4 + p) * 2 + 0];
      float oy = offp[(l * 4 + p) * 2 + 1];
      float aw = awp[l * 4 + p];
      // loc = ref.xy + off/P * ref.zw * 0.5 ; ix = loc*W - 0.5 (align_corners=False)
      float ix = (rp.x + ox * 0.125f * rp.z) * (float)Wl - 0.5f;
      float iy = (rp.y + oy * 0.125f * rp.w) * (float)Hl - 0.5f;
      float x0f = floorf(ix), y0f = floorf(iy);
      float wx1 = ix - x0f,  wy1 = iy - y0f;
      float wx0 = 1.0f - wx1, wy0 = 1.0f - wy1;
      int x0 = (int)x0f, y0 = (int)y0f;
      int x1 = x0 + 1,   y1 = y0 + 1;
      bool vx0 = (x0 >= 0) && (x0 < Wl);
      bool vx1 = (x1 >= 0) && (x1 < Wl);
      if (y0 >= 0 && y0 < Hl) {
        const float* rowp = vb + (size_t)(s0 + y0 * Wl) * DIM;
        if (vx0) acc += aw * wx0 * wy0 * rowp[(size_t)x0 * DIM];
        if (vx1) acc += aw * wx1 * wy0 * rowp[(size_t)x1 * DIM];
      }
      if (y1 >= 0 && y1 < Hl) {
        const float* rowp = vb + (size_t)(s0 + y1 * Wl) * DIM;
        if (vx0) acc += aw * wx0 * wy1 * rowp[(size_t)x0 * DIM];
        if (vx1) acc += aw * wx1 * wy1 * rowp[(size_t)x1 * DIM];
      }
    }
  }
  out[(size_t)bq * DIM + h * 32 + lane] = acc;
}

// =====================================================================
extern "C" void kernel_launch(void* const* d_in, const int* in_sizes, int n_in,
                              void* d_out, int out_size, void* d_ws, size_t ws_size,
                              hipStream_t stream)
{
  (void)in_sizes; (void)n_in; (void)out_size; (void)ws_size;
  const float* query  = (const float*)d_in[0];   // [16,300,256]
  const float* refp   = (const float*)d_in[1];   // [16,300,4]
  const float* value  = (const float*)d_in[2];   // [16,8400,256]
  const float* W_off  = (const float*)d_in[3];   // [256,192]
  const float* b_off  = (const float*)d_in[4];   // [192]
  const float* W_attn = (const float*)d_in[5];   // [256,96]
  const float* b_attn = (const float*)d_in[6];   // [96]
  const float* W_val  = (const float*)d_in[7];   // [256,256]
  const float* b_val  = (const float*)d_in[8];   // [256]
  const float* W_out  = (const float*)d_in[9];   // [256,256]
  const float* b_out  = (const float*)d_in[10];  // [256]
  float* out = (float*)d_out;                    // [16,300,256]

  const int BQ = BATCH * LQ;                     // 4800
  const int MV = BATCH * LV;                     // 134400

  float* ws     = (float*)d_ws;
  float* vproj  = ws;                                   // 134400*256
  float* offb   = vproj + (size_t)MV * DIM;             // 4800*192
  float* attnb  = offb  + (size_t)BQ * 192;             // 4800*96
  float* interm = attnb + (size_t)BQ * 96;              // 4800*256

  dim3 blk(256);

  // 1) value projection (memory-bound GEMM; X read exactly once)
  gemm_bias_wmma<<<dim3(1, MV / 128), blk, 0, stream>>>(
      value, W_val, b_val, vproj, MV, DIM);

  // 2) sampling offsets
  gemm_bias_wmma<<<dim3(1, (BQ + 127) / 128), blk, 0, stream>>>(
      query, W_off, b_off, offb, BQ, 192);

  // 3) attention logits
  gemm_bias_wmma<<<dim3(1, (BQ + 127) / 128), blk, 0, stream>>>(
      query, W_attn, b_attn, attnb, BQ, 96);

  // 4) softmax over 12 per (b,q,h)
  int total = BQ * NUM_HEADS;                    // 38400
  softmax12_kernel<<<(total + 255) / 256, blk, 0, stream>>>(attnb, total);

  // 5) deformable sampling + aggregation (wave per (b,q,h))
  ms_deform_sample_kernel<<<total / 8, blk, 0, stream>>>(
      vproj, refp, offb, attnb, interm);

  // 6) output projection
  gemm_bias_wmma<<<dim3(1, (BQ + 127) / 128), blk, 0, stream>>>(
      interm, W_out, b_out, out, BQ, DIM);
}
